// WLA_43774306681332
// MI455X (gfx1250) — matrix operand order repaired and checked by reading, not compiled, not run on hardware
//
#include <hip/hip_runtime.h>
#include <hip/hip_bf16.h>

// ---------- types ----------
typedef __bf16 bf16_t;
typedef bf16_t v16bf __attribute__((ext_vector_type(16)));
typedef bf16_t v8bf  __attribute__((ext_vector_type(8)));
typedef float  v8f   __attribute__((ext_vector_type(8)));

// ---------- problem constants ----------
#define B_    8
#define C_    192
#define HW_   224
#define P_    7
#define PP_   49        // tokens per window
#define NH_   6
#define HD_   32
#define MT_   64        // padded token dim (49 -> 64)
#define NQKV_ (3*C_)    // 576
#define SCALE_ 0.17677669529663687f   // 1/sqrt(32)

// workspace layout (bytes)
#define WS_WQKV  0            // 576*192 bf16 = 221184 B
#define WS_WOUT  221184       // 192*192 bf16 =  73728 B
#define WS_BIAS  294912       // 6*49*49 f32  =  57624 B

__device__ __forceinline__ bf16_t tobf(float f) {
  unsigned u = __builtin_bit_cast(unsigned, f);
  u = (u + 0x7fffu + ((u >> 16) & 1u)) >> 16;     // round-to-nearest-even
  unsigned short s = (unsigned short)u;
  return __builtin_bit_cast(bf16_t, s);
}

__device__ __forceinline__ v16bf catv(v8bf lo, v8bf hi) {
  return __builtin_shufflevector(lo, hi, 0,1,2,3,4,5,6,7,8,9,10,11,12,13,14,15);
}

// A fragment (16x32 bf16): lane l holds row M; K chunks at kb+8*half and kb+16+8*half
__device__ __forceinline__ v16bf loadA(const bf16_t* rowp, int kb, int half) {
  v8bf lo = *(const v8bf*)(rowp + kb + 8*half);
  v8bf hi = *(const v8bf*)(rowp + kb + 16 + 8*half);
  return catv(lo, hi);
}
// B fragment (32x16 bf16): lane l holds col N; K chunks at kb+8*half and kb+16+8*half
__device__ __forceinline__ v16bf loadB(const bf16_t* colp, int kb, int half) {
  v8bf lo = *(const v8bf*)(colp + kb + 8*half);
  v8bf hi = *(const v8bf*)(colp + kb + 16 + 8*half);
  return catv(lo, hi);
}

// ---------- prep: weights -> bf16, expand rel-pos bias table ----------
__global__ void WLA_prep(const float* __restrict__ wqkv, const float* __restrict__ wout,
                         const float* __restrict__ rel,
                         bf16_t* __restrict__ wqkv_bf, bf16_t* __restrict__ wout_bf,
                         float* __restrict__ biasTab) {
  int i = blockIdx.x * 256 + threadIdx.x;
  const int N1 = NQKV_ * C_;       // 110592
  const int N2 = C_ * C_;          // 36864
  const int N3 = NH_ * PP_ * PP_;  // 14406
  if (i < N1) { wqkv_bf[i] = tobf(wqkv[i]); return; }
  i -= N1;
  if (i < N2) { wout_bf[i] = tobf(wout[i]); return; }
  i -= N2;
  if (i < N3) {
    int h = i / (PP_*PP_);
    int r = i - h * (PP_*PP_);
    int a = r / PP_, b = r - a * PP_;
    int ri = a / P_ - b / P_ + (P_ - 1);
    int rj = a % P_ - b % P_ + (P_ - 1);
    biasTab[i] = rel[(h * 13 + ri) * 13 + rj];
  }
}

// ---------- fused per-window attention ----------
__global__ __launch_bounds__(256) void WLA_attn(
    const float* __restrict__ x,
    const float* __restrict__ bqkv,
    const float* __restrict__ bout,
    const bf16_t* __restrict__ wqkv_bf,
    const bf16_t* __restrict__ wout_bf,
    const float* __restrict__ biasTab,
    float* __restrict__ out) {

  const int tid  = threadIdx.x;
  const int wave = tid >> 5;
  const int lane = tid & 31;
  const int half = lane >> 4;
  const int l    = lane & 15;

  const int wid = blockIdx.x;          // 0..8191
  const int bi  = wid >> 10;
  const int w   = wid & 1023;
  const int wy  = w >> 5, wx = w & 31;
  const int h0  = wy * P_, w0 = wx * P_;

  __shared__ __align__(128) unsigned char smem[122880];   // 120 KB
  bf16_t* Xs   = (bf16_t*)smem;                 // [64][192] bf16, 24KB (stage 1-2)
  float*  Sbuf = (float*)smem;                  // [64][64] f32, 16KB (stage 3, reuses Xs)
  bf16_t* Pbuf = (bf16_t*)(smem + 16384);       // [64][64] bf16, 8KB (stage 3)
  bf16_t* Qs   = (bf16_t*)(smem + 24576);       // [64][192] bf16, 24KB
  bf16_t* Ks   = (bf16_t*)(smem + 49152);       // [64][192] bf16, 24KB
  float*  Ybuf = (float*)(smem + 24576);        // [64][192] f32, 48KB (stage 4, reuses Qs+Ks)
  bf16_t* Vt   = (bf16_t*)(smem + 73728);       // [192][64] bf16, 24KB (V transposed)
  bf16_t* Os   = (bf16_t*)(smem + 98304);       // [64][192] bf16, 24KB

  const int mt = wave & 3;             // per-wave constant M-tile (stride-8 schedule)
  const int ntBase = wave >> 2;        // starting N-tile, step 2

  // --- stage 1: gather window tokens, fp32 -> bf16, channel-major global reads ---
  for (int idx = tid; idx < C_ * PP_; idx += 256) {
    int ch = idx / PP_, tok = idx - ch * PP_;
    int ti = tok / P_, tj = tok - ti * P_;
    float v = x[(((bi * C_ + ch) * HW_) + h0 + ti) * HW_ + (w0 + tj)];
    Xs[tok * C_ + ch] = tobf(v);
  }
  for (int idx = tid; idx < (MT_ - PP_) * C_; idx += 256)
    Xs[PP_ * C_ + idx] = tobf(0.0f);
  __syncthreads();

  // --- stage 2: QKV GEMM  (64x192) @ (192x576) ---
  {
    // A fragments are per-wave invariant: load the whole 16x192 slab once (48 VGPRs)
    const bf16_t* arow = Xs + (mt * 16 + l) * C_;
    v16bf afrag[6];
#pragma unroll
    for (int ks = 0; ks < 6; ++ks) afrag[ks] = loadA(arow, ks * 32, half);

    for (int nt = ntBase; nt < 36; nt += 2) {
      v8f acc = {};
      const bf16_t* brow = wqkv_bf + (nt * 16 + l) * C_;
#pragma unroll
      for (int ks = 0; ks < 6; ++ks) {
        v16bf b = loadB(brow, ks * 32, half);
        acc = __builtin_amdgcn_wmma_f32_16x16x32_bf16(false, afrag[ks], false, b,
                                                      (short)0, acc, false, false);
      }
      float bias = bqkv[nt * 16 + l];
      int s  = nt / 12;                  // uniform across lanes: q/k/v select
      int ch = (nt - s * 12) * 16 + l;
      if (s == 0) {
#pragma unroll
        for (int r = 0; r < 8; ++r)
          Qs[(mt * 16 + r + 8 * half) * C_ + ch] = tobf((acc[r] + bias) * SCALE_);
      } else if (s == 1) {
#pragma unroll
        for (int r = 0; r < 8; ++r)
          Ks[(mt * 16 + r + 8 * half) * C_ + ch] = tobf(acc[r] + bias);
      } else {
#pragma unroll
        for (int r = 0; r < 8; ++r)
          Vt[ch * MT_ + (mt * 16 + r + 8 * half)] = tobf(acc[r] + bias);
      }
    }
  }
  __syncthreads();

  // --- stage 3: per-head attention ---
  for (int h = 0; h < NH_; ++h) {
    // S = Q_h @ K_h^T  (64x64, K=32): 16 tiles; A invariant per wave
    {
      v16bf a = loadA(Qs + (mt * 16 + l) * C_ + h * HD_, 0, half);
#pragma unroll
      for (int u = 0; u < 2; ++u) {
        int nt = ntBase + 2 * u;
        v8f acc = {};
        v16bf b = loadB(Ks + (nt * 16 + l) * C_ + h * HD_, 0, half);
        acc = __builtin_amdgcn_wmma_f32_16x16x32_bf16(false, a, false, b,
                                                      (short)0, acc, false, false);
        int j  = nt * 16 + l;
        int jc = (j < PP_) ? j : (PP_ - 1);
#pragma unroll
        for (int r = 0; r < 8; ++r) {
          int i  = mt * 16 + r + 8 * half;
          int ic = (i < PP_) ? i : (PP_ - 1);
          float bv = biasTab[(h * PP_ + ic) * PP_ + jc];      // unconditional load
          float v  = (i < PP_ && j < PP_) ? (acc[r] + bv) : -1e30f;
          Sbuf[i * MT_ + j] = v;
        }
      }
    }
    __syncthreads();
    // softmax: one thread per row
    if (tid < MT_) {
      float mx = -1e30f;
      for (int j = 0; j < MT_; ++j) mx = fmaxf(mx, Sbuf[tid * MT_ + j]);
      float sum = 0.f;
      for (int j = 0; j < MT_; ++j) {
        float e = __expf(Sbuf[tid * MT_ + j] - mx);
        Sbuf[tid * MT_ + j] = e;
        sum += e;
      }
      float inv = 1.0f / sum;
      for (int j = 0; j < MT_; ++j)
        Pbuf[tid * MT_ + j] = tobf(Sbuf[tid * MT_ + j] * inv);
    }
    __syncthreads();
    // O_h = P @ V_h  (64x32, K=64): 8 tiles -> one per wave
    {
      int nt = wave >> 2;   // 0..1
      v8f acc = {};
      const bf16_t* arow = Pbuf + (mt * 16 + l) * MT_;
      const bf16_t* brow = Vt + (h * HD_ + nt * 16 + l) * MT_;
#pragma unroll
      for (int ks = 0; ks < 2; ++ks) {
        v16bf a = loadA(arow, ks * 32, half);
        v16bf b = loadB(brow, ks * 32, half);
        acc = __builtin_amdgcn_wmma_f32_16x16x32_bf16(false, a, false, b,
                                                      (short)0, acc, false, false);
      }
      int ch = h * HD_ + nt * 16 + l;
#pragma unroll
      for (int r = 0; r < 8; ++r)
        Os[(mt * 16 + r + 8 * half) * C_ + ch] = tobf(acc[r]);
    }
    __syncthreads();
  }

  // --- stage 4: output projection (64x192)@(192x192): A invariant per wave ---
  {
    const bf16_t* arow = Os + (mt * 16 + l) * C_;
    v16bf afrag[6];
#pragma unroll
    for (int ks = 0; ks < 6; ++ks) afrag[ks] = loadA(arow, ks * 32, half);

    for (int nt = ntBase; nt < 12; nt += 2) {
      v8f acc = {};
      const bf16_t* brow = wout_bf + (nt * 16 + l) * C_;
#pragma unroll
      for (int ks = 0; ks < 6; ++ks) {
        v16bf b = loadB(brow, ks * 32, half);
        acc = __builtin_amdgcn_wmma_f32_16x16x32_bf16(false, afrag[ks], false, b,
                                                      (short)0, acc, false, false);
      }
      int n = nt * 16 + l;
      float bias = bout[n];
#pragma unroll
      for (int r = 0; r < 8; ++r)
        Ybuf[(mt * 16 + r + 8 * half) * C_ + n] = acc[r] + bias;
    }
  }
  __syncthreads();

  // --- stage 5: scatter back to NCHW fp32 ---
  for (int idx = tid; idx < C_ * PP_; idx += 256) {
    int ch = idx / PP_, tok = idx - ch * PP_;
    int ti = tok / P_, tj = tok - ti * P_;
    out[(((bi * C_ + ch) * HW_) + h0 + ti) * HW_ + (w0 + tj)] = Ybuf[tok * C_ + ch];
  }
}

extern "C" void kernel_launch(void* const* d_in, const int* in_sizes, int n_in,
                              void* d_out, int out_size, void* d_ws, size_t ws_size,
                              hipStream_t stream) {
  const float* x    = (const float*)d_in[0];
  const float* wqkv = (const float*)d_in[1];
  const float* bqkv = (const float*)d_in[2];
  const float* rel  = (const float*)d_in[3];
  const float* wout = (const float*)d_in[4];
  const float* bout = (const float*)d_in[5];
  float* out = (float*)d_out;

  bf16_t* wqkv_bf = (bf16_t*)((char*)d_ws + WS_WQKV);
  bf16_t* wout_bf = (bf16_t*)((char*)d_ws + WS_WOUT);
  float*  biasTab = (float*)((char*)d_ws + WS_BIAS);

  const int prep_n = NQKV_ * C_ + C_ * C_ + NH_ * PP_ * PP_;  // 161862
  WLA_prep<<<(prep_n + 255) / 256, 256, 0, stream>>>(wqkv, wout, rel,
                                                     wqkv_bf, wout_bf, biasTab);

  const int nWin = B_ * 32 * 32;  // 8192 windows
  WLA_attn<<<nWin, 256, 0, stream>>>(x, bqkv, bout, wqkv_bf, wout_bf, biasTab, out);
}
